// SSKernelDiag_59158879535426
// MI455X (gfx1250) — matrix operand order repaired and checked by reading, not compiled, not run on hardware
//
#include <hip/hip_runtime.h>
#include <hip/hip_bf16.h>

// S4D Vandermonde kernel generation, refactored as per-head 64x64x128 bf16 GEMM.
//   out[h, 64*l1 + l0] = sum_k M[l1,k] * N[k,l0]
//   M[l1,n]=Re(C'A^{64l1}), M[l1,64+n]=-Im(C'A^{64l1}); N[n,l0]=Re(A^{l0}), N[64+n,l0]=Im(A^{l0})
// H=256, N=64, L=4096 (reference constants; L input is not read; structure is fixed).

typedef __bf16 v16bf __attribute__((ext_vector_type(16)));
typedef float  v8f   __attribute__((ext_vector_type(8)));

#define NH   256
#define NN   64
#define NL   4096
#define STR  136          // padded LDS row stride in bf16 elems (272 B, 16B-aligned, bank-skewed)

union FragU { uint4 q[2]; v16bf v; };

__device__ __forceinline__ unsigned short f2bf(float x) {
    unsigned int b = __float_as_uint(x);
    b += 0x7fffu + ((b >> 16) & 1u);          // round-to-nearest-even
    return (unsigned short)(b >> 16);
}

__global__ __launch_bounds__(128) void ssd_vandermonde_wmma(
    const float* __restrict__ A_abslog,   // (H, N)
    const float* __restrict__ A_phase,    // (H, N)
    const float* __restrict__ C,          // (1, H, N, 2)
    float* __restrict__ out)              // (1, H, L)
{
    __shared__ float sAbs[NN], sPh[NN], sCr[NN], sCi[NN];
    __shared__ unsigned short Mlds[64 * STR];   // M[l1][k], k = 0..127
    __shared__ unsigned short Tlds[64 * STR];   // N^T: T[l0][k]

    const int h   = blockIdx.x;
    const int tid = threadIdx.x;

    // ---- Stage 1: per-eigenvalue C' = 2 * C * (A-1)/log(A) -------------------
    if (tid < NN) {
        const int n  = tid;
        const float al = A_abslog[h * NN + n];
        const float ph = A_phase [h * NN + n];
        const float r  = __expf(al);
        float sn, cs;  __sincosf(ph, &sn, &cs);
        const float Ar = r * cs, Ai = r * sn;
        const float inv = 1.0f / (al * al + ph * ph);   // al < 0 strictly => denom > 0
        const float rr = ((Ar - 1.0f) * al + Ai * ph) * inv;   // Re[(A-1)/dtA]
        const float ri = (Ai * al - (Ar - 1.0f) * ph) * inv;   // Im[(A-1)/dtA]
        const float Cr = C[(h * NN + n) * 2 + 0];
        const float Ci = C[(h * NN + n) * 2 + 1];
        sCr[n] = 2.0f * (Cr * rr - Ci * ri);            // fold the final *2 here
        sCi[n] = 2.0f * (Cr * ri + Ci * rr);
        sAbs[n] = al;  sPh[n] = ph;
    }
    __syncthreads();

    // ---- Stage 2: build M (B = C'*Q, Q = A^{64*l1}) and N^T (P = A^{l0}) -----
    for (int idx = tid; idx < 64 * 64; idx += 128) {
        const int li = idx >> 6;          // doubles as l1 and l0
        const int n  = idx & 63;
        const float al = sAbs[n], ph = sPh[n];

        // Q = A^{64*li}; B = C' * Q
        {
            const float t = (float)(li << 6);
            const float mag = __expf(t * al);
            float s, c;  __sincosf(t * ph, &s, &c);
            const float Qr = mag * c, Qi = mag * s;
            const float Br = sCr[n] * Qr - sCi[n] * Qi;
            const float Bi = sCr[n] * Qi + sCi[n] * Qr;
            Mlds[li * STR + n]      = f2bf(Br);
            Mlds[li * STR + 64 + n] = f2bf(-Bi);
        }
        // P = A^{li}
        {
            const float t = (float)li;
            const float mag = __expf(t * al);
            float s, c;  __sincosf(t * ph, &s, &c);
            Tlds[li * STR + n]      = f2bf(mag * c);    // Pr
            Tlds[li * STR + 64 + n] = f2bf(mag * s);    // Pi
        }
    }
    __syncthreads();

    // ---- Stage 3: 64x64x128 GEMM via v_wmma_f32_16x16x32_bf16 ----------------
    const int wave = tid >> 5;
    const int lane = tid & 31;
    const int m0   = wave << 4;                 // this wave's row tile (l1 block)
    const int rowA = m0 + (lane & 15);
    const int colB = lane & 15;
    const int hi8  = (lane >> 4) << 3;          // 0 / 8  (A-frag K skew)
    const int hi16 = (lane >> 4) << 4;          // 0 / 16 (unused)

    v8f acc[4] = {};

    #pragma unroll
    for (int kc = 0; kc < 4; ++kc) {
        const int k0 = kc << 5;

        // A fragment (ISA 16-bit A 16x32 layout): halves 0..7 = K {k0+hi8 ..+7},
        // halves 8..15 = K {k0+16+hi8 ..+7}
        FragU a;
        a.q[0] = *(const uint4*)&Mlds[rowA * STR + k0 + hi8];
        a.q[1] = *(const uint4*)&Mlds[rowA * STR + k0 + 16 + hi8];

        #pragma unroll
        for (int nt = 0; nt < 4; ++nt) {
            // B fragment: column = lane&15, halves 0..7 = K {k0+hi8 ..+7}, halves 8..15 = K {k0+16+hi8 ..+7}
            const unsigned short* bp = &Tlds[(nt * 16 + colB) * STR + k0 + hi8];
            FragU b;
            b.q[0] = *(const uint4*)bp;
            b.q[1] = *(const uint4*)(bp + 16);

            acc[nt] = __builtin_amdgcn_wmma_f32_16x16x32_bf16(
                false, a.v, false, b.v, (short)0, acc[nt], false, false);
        }
    }

    // ---- Store: C/D layout -> out[h, (l1)*64 + l0] ---------------------------
    float* outp = out + h * NL;
    const int rowBase = m0 + ((lane >> 4) << 3);   // VGPR r: row = r (+8 for hi lanes)
    const int col     = lane & 15;
    #pragma unroll
    for (int nt = 0; nt < 4; ++nt) {
        #pragma unroll
        for (int r = 0; r < 8; ++r) {
            outp[(rowBase + r) * 64 + nt * 16 + col] = acc[nt][r];
        }
    }
}

extern "C" void kernel_launch(void* const* d_in, const int* in_sizes, int n_in,
                              void* d_out, int out_size, void* d_ws, size_t ws_size,
                              hipStream_t stream) {
    const float* A_abslog = (const float*)d_in[0];   // H*N
    const float* A_phase  = (const float*)d_in[1];   // H*N
    const float* C        = (const float*)d_in[2];   // 1*H*N*2
    // d_in[3] = L (scalar, device mem); structure fixed at L=4096 per reference.
    float* out = (float*)d_out;                      // 1*H*L

    ssd_vandermonde_wmma<<<NH, 128, 0, stream>>>(A_abslog, A_phase, C, out);
}
